// GAT_1700807049283
// MI455X (gfx1250) — matrix operand order repaired and checked by reading, not compiled, not run on hardware
//
#include <hip/hip_runtime.h>
#include <stdint.h>

typedef __attribute__((ext_vector_type(16))) __bf16 v16bf;
typedef __attribute__((ext_vector_type(8)))  float  v8f;

#define NN 50000
#define NE 400000

// ---------- helpers ----------
__device__ __forceinline__ uint16_t f2bf(float f) {          // RNE f32 -> bf16 bits
  uint32_t u = __float_as_uint(f);
  u += 0x7FFFu + ((u >> 16) & 1u);
  return (uint16_t)(u >> 16);
}
__device__ __forceinline__ unsigned encf(float f) {          // order-preserving f32 -> u32
  unsigned u = __float_as_uint(f);
  return (u & 0x80000000u) ? ~u : (u | 0x80000000u);
}
__device__ __forceinline__ float decf(unsigned u) {
  unsigned v = (u & 0x80000000u) ? (u & 0x7FFFFFFFu) : ~u;
  return __uint_as_float(v);
}

union U4BF { uint4 u; __bf16 h[8]; };

// async copy of 16B global -> LDS (ASYNCcnt-tracked), per CDNA5 ISA.
// lds_base is passed as an extra unused operand so the LDS object escapes into the asm
// (otherwise LLVM proves the shared array is never written and folds ds_loads to undef).
__device__ __forceinline__ void async_g2l_b128(uint32_t lds_byte_off, const void* gaddr,
                                               const void* lds_base) {
  asm volatile("global_load_async_to_lds_b128 %0, %1, off"
               :: "v"(lds_byte_off), "v"(gaddr), "v"(lds_base) : "memory");
}
__device__ __forceinline__ void wait_async0() {
  asm volatile("s_wait_asynccnt 0x0" ::: "memory");
}

// ---------- WMMA GEMM with async-LDS B panel ----------
// C[M,N] = A[M,K] @ B[K,N] (+bias). Bt is B transposed: [N,K] bf16.
// Block = 8 waves -> supertile 128(M) x 64(N). Each wave: 16x64 (4 accumulators).
// Stage = 64 K: B panel (64 cols x 64 K) double-buffered in LDS via async copies
// (2 x 16B per thread per fill). LDS col stride = 72 halfwords (144B): 16B-aligned,
// bank-conflict-free (gcd(36,64)=4 -> 16 distinct banks across 16 lanes).
// 8 WMMAs per barrier round; K must be a multiple of 64.
__global__ __launch_bounds__(256) void k_gemm_lds(
    const uint16_t* __restrict__ A, const uint16_t* __restrict__ Bt,
    float* __restrict__ C, uint16_t* __restrict__ Cbf,
    const float* __restrict__ bias,
    int M, int N, int K, int tilesM)
{
  __shared__ __align__(16) uint16_t sB[2 * 64 * 72];

  const int tid  = threadIdx.x;
  const int wv   = tid >> 5;
  const int lane = tid & 31;
  const int half = lane >> 4;
  const int l15  = lane & 15;

  int tm = blockIdx.x * 8 + wv;
  if (tm >= tilesM) tm = tilesM - 1;            // clamp: keep EXEC all-1, guard stores
  const int colBase = blockIdx.y * 64;

  // A fragment source: lane row = tm*16+l15; element e -> K = k0 + (e/8)*16 + half*8 + e%8
  const uint16_t* arow = A + (size_t)(tm * 16 + l15) * K + half * 8;

  // async fill mapping: thread t copies 2 x 16B chunks (col = t>>2, k-chunks t&3 and (t&3)+4)
  const int fcol = tid >> 2, fch = tid & 3;
  const uint16_t* bsrc = Bt + (size_t)(colBase + fcol) * K + fch * 8;
  const uint32_t lds_off0 = (uint32_t)(fcol * 72 + fch * 8) * 2;   // bytes, buffer 0
  const uint32_t bufBytes = 64u * 72u * 2u;

  // prologue: fill buffer 0 (k = 0..63)
  async_g2l_b128(lds_off0, bsrc, sB);
  async_g2l_b128(lds_off0 + 64, bsrc + 32, sB);
  wait_async0();
  __syncthreads();

  v8f z = {0.f, 0.f, 0.f, 0.f, 0.f, 0.f, 0.f, 0.f};
  v8f acc[4] = {z, z, z, z};

  const int stages = K / 64;
  for (int s = 0; s < stages; ++s) {
    const int cur = s & 1;
    if (s + 1 < stages) {
      const uint16_t* nb = bsrc + (size_t)(s + 1) * 64;
      const uint32_t noff = lds_off0 + (cur ^ 1) * bufBytes;
      async_g2l_b128(noff, nb, sB);
      async_g2l_b128(noff + 64, nb + 32, sB);
      __builtin_prefetch(arow + (size_t)(s + 1) * 64, 0, 3);   // global_prefetch_b8
    }

    // hoist both A fragments of this stage (4 x global b128)
    U4BF alo[2], ahi[2];
#pragma unroll
    for (int ks = 0; ks < 2; ++ks) {
      const size_t kofs = (size_t)s * 64 + ks * 32;
      alo[ks].u = *(const uint4*)(arow + kofs);
      ahi[ks].u = *(const uint4*)(arow + kofs + 16);
    }

    const uint16_t* sb = &sB[cur * 64 * 72];
#pragma unroll
    for (int ks = 0; ks < 2; ++ks) {
      v16bf a;
#pragma unroll
      for (int i = 0; i < 8; ++i) { a[i] = alo[ks].h[i]; a[i + 8] = ahi[ks].h[i]; }

      // batch all 4 B fragment pairs (8 ds_load_b128) before the WMMAs
      U4BF blo[4], bhi[4];
#pragma unroll
      for (int t = 0; t < 4; ++t) {
        const uint16_t* bp = sb + (t * 16 + l15) * 72 + ks * 32 + half * 8;
        blo[t].u = *(const uint4*)(bp);
        bhi[t].u = *(const uint4*)(bp + 16);
      }
#pragma unroll
      for (int t = 0; t < 4; ++t) {
        v16bf b;
#pragma unroll
        for (int i = 0; i < 8; ++i) { b[i] = blo[t].h[i]; b[i + 8] = bhi[t].h[i]; }
        acc[t] = __builtin_amdgcn_wmma_f32_16x16x32_bf16(false, a, false, b,
                                                         (short)0, acc[t], false, false);
      }
    }
    wait_async0();        // next buffer resident
    __syncthreads();      // everyone done before next fill overwrites
  }

  // store 16x64 per wave; C layout: VGPR r -> row = half*8 + r, col = lane&15
#pragma unroll
  for (int t = 0; t < 4; ++t) {
    int col = colBase + t * 16 + l15;
    float bb = bias ? bias[col] : 0.f;
#pragma unroll
    for (int r = 0; r < 8; ++r) {
      int row = tm * 16 + half * 8 + r;
      if (row < M) {
        float v = acc[t][r] + bb;
        C[(size_t)row * N + col] = v;
        if (Cbf) Cbf[(size_t)row * N + col] = f2bf(v);
      }
    }
  }
}

// ---------- simple WMMA GEMM for narrow N (N = 16) ----------
__global__ __launch_bounds__(256) void k_gemm_n16(
    const uint16_t* __restrict__ A, const uint16_t* __restrict__ Bt,
    float* __restrict__ C, int M, int K, int tilesM)
{
  int wave = blockIdx.x * (blockDim.x >> 5) + (threadIdx.x >> 5);
  if (wave >= tilesM) return;
  int lane = threadIdx.x & 31;
  int half = lane >> 4, l15 = lane & 15;
  const uint16_t* arow = A + (size_t)(wave * 16 + l15) * K + half * 8;
  const uint16_t* brow = Bt + (size_t)l15 * K + half * 8;
  v8f acc = {0.f, 0.f, 0.f, 0.f, 0.f, 0.f, 0.f, 0.f};
  for (int k0 = 0; k0 < K; k0 += 32) {
    U4BF alo, ahi, blo, bhi;
    alo.u = *(const uint4*)(arow + k0);
    ahi.u = *(const uint4*)(arow + k0 + 16);
    blo.u = *(const uint4*)(brow + k0);
    bhi.u = *(const uint4*)(brow + k0 + 16);
    v16bf a, b;
#pragma unroll
    for (int i = 0; i < 8; ++i) {
      a[i] = alo.h[i]; a[i + 8] = ahi.h[i];
      b[i] = blo.h[i]; b[i + 8] = bhi.h[i];
    }
    acc = __builtin_amdgcn_wmma_f32_16x16x32_bf16(false, a, false, b,
                                                  (short)0, acc, false, false);
  }
#pragma unroll
  for (int r = 0; r < 8; ++r) {
    int row = wave * 16 + half * 8 + r;
    if (row < M) C[(size_t)row * 16 + l15] = acc[r];
  }
}

// ---------- utility kernels ----------
__global__ void k_cvt_bf16_4(const float4* __restrict__ in, ushort4* __restrict__ out,
                             long long n4) {
  long long i = (long long)blockIdx.x * blockDim.x + threadIdx.x;
  if (i >= n4) return;
  float4 v = in[i];
  out[i] = make_ushort4(f2bf(v.x), f2bf(v.y), f2bf(v.z), f2bf(v.w));
}
__global__ void k_transpose_bf16(const float* __restrict__ W, uint16_t* __restrict__ Wt,
                                 int K, int N) {              // W[K,N] -> Wt[N,K] bf16
  int i = blockIdx.x * blockDim.x + threadIdx.x;
  if (i >= K * N) return;
  int n = i / K, k = i % K;
  Wt[i] = f2bf(W[(size_t)k * N + n]);
}
__global__ void k_zero_f32_4(float4* __restrict__ p, long long n4) {
  long long i = (long long)blockIdx.x * blockDim.x + threadIdx.x;
  if (i < n4) p[i] = make_float4(0.f, 0.f, 0.f, 0.f);
}
__global__ void k_init_attn(unsigned* __restrict__ mbuf, float* __restrict__ sbuf, int n) {
  int i = blockIdx.x * blockDim.x + threadIdx.x;
  if (i < n) { mbuf[i] = 0x007FFFFFu; sbuf[i] = 0.f; }        // enc(-inf)
}

// per-node, per-head attention scores; al/ar staged in LDS
__global__ __launch_bounds__(256) void k_scores(
    const float* __restrict__ ft, const float* __restrict__ al, const float* __restrict__ ar,
    float* __restrict__ el, float* __restrict__ er, int Nn, int H, int D) {
  __shared__ __align__(16) float sal[512], sar[512];
  int HD = H * D;
  for (int i = threadIdx.x; i < HD; i += 256) { sal[i] = al[i]; sar[i] = ar[i]; }
  __syncthreads();
  int idx = blockIdx.x * 256 + threadIdx.x;
  if (idx >= Nn * H) return;
  int n = idx / H, h = idx % H;
  const float4* f = (const float4*)(ft + (size_t)n * HD + h * D);
  const float4* a = (const float4*)(sal + h * D);
  const float4* b = (const float4*)(sar + h * D);
  float sl = 0.f, sr = 0.f;
  for (int d = 0; d < D / 4; ++d) {
    float4 fv = f[d], av = a[d], bv = b[d];
    sl += fv.x * av.x + fv.y * av.y + fv.z * av.z + fv.w * av.w;
    sr += fv.x * bv.x + fv.y * bv.y + fv.z * bv.z + fv.w * bv.w;
  }
  el[idx] = sl; er[idx] = sr;
}

// pass 1: per-edge logits (all heads) + segment-max via ordered-uint atomicMax
__global__ void k_edge_max(const long long* __restrict__ src, const long long* __restrict__ dst,
                           const float* __restrict__ el, const float* __restrict__ er,
                           float* __restrict__ ebuf, unsigned* __restrict__ mbuf, int E, int H) {
  int e = blockIdx.x * blockDim.x + threadIdx.x;
  if (e >= E) return;
  long long s = src[e], d = dst[e];
  for (int h = 0; h < H; ++h) {
    float x = el[s * H + h] + er[d * H + h];
    x = x > 0.f ? x : 0.2f * x;                               // leaky_relu(0.2)
    ebuf[(size_t)e * H + h] = x;
    atomicMax(&mbuf[d * H + h], encf(x));
  }
}

// pass 2: exp(e - m) + segment-sum
__global__ void k_edge_exp(const long long* __restrict__ dst, float* __restrict__ ebuf,
                           const unsigned* __restrict__ mbuf, float* __restrict__ sbuf,
                           int E, int H) {
  int e = blockIdx.x * blockDim.x + threadIdx.x;
  if (e >= E) return;
  long long d = dst[e];
  for (int h = 0; h < H; ++h) {
    float m = decf(mbuf[d * H + h]);
    if (!isfinite(m)) m = 0.f;
    float ex = expf(ebuf[(size_t)e * H + h] - m);
    ebuf[(size_t)e * H + h] = ex;
    atomicAdd(&sbuf[d * H + h], ex);
  }
}

// pass 3: alpha-weighted scatter-add of ft[src] into agg[dst], float4 gathers
__global__ void k_edge_aggr(const long long* __restrict__ src, const long long* __restrict__ dst,
                            const float* __restrict__ ft, const float* __restrict__ ebuf,
                            const float* __restrict__ sbuf, float* __restrict__ agg,
                            long long total4, int H, int D) {
  long long i = (long long)blockIdx.x * blockDim.x + threadIdx.x;
  if (i >= total4) return;
  int q4 = (H * D) >> 2;
  long long e = i / q4;
  int q = (int)(i % q4);
  int h = q / (D >> 2);
  long long s = src[e], d = dst[e];
  float coeff = ebuf[e * H + h] / (sbuf[d * H + h] + 1e-16f);
  float4 v = ((const float4*)ft)[s * q4 + q];
  float* p = agg + (size_t)d * H * D + (size_t)q * 4;
  atomicAdd(p + 0, v.x * coeff);
  atomicAdd(p + 1, v.y * coeff);
  atomicAdd(p + 2, v.z * coeff);
  atomicAdd(p + 3, v.w * coeff);
}

// epilogue: + residual + bias, optional ELU, f32 out + optional bf16 recast (vectorized)
__global__ void k_finalize4(const float4* __restrict__ agg, const float4* __restrict__ res,
                            const float4* __restrict__ bias, float4* __restrict__ out,
                            ushort4* __restrict__ outbf, long long total4, int HD4, int act) {
  long long i = (long long)blockIdx.x * blockDim.x + threadIdx.x;
  if (i >= total4) return;
  int c = (int)(i % HD4);
  float4 v = agg[i];
  float4 bv = bias[c];
  v.x += bv.x; v.y += bv.y; v.z += bv.z; v.w += bv.w;
  if (res) { float4 r = res[i]; v.x += r.x; v.y += r.y; v.z += r.z; v.w += r.w; }
  if (act) {
    v.x = v.x > 0.f ? v.x : expm1f(v.x);
    v.y = v.y > 0.f ? v.y : expm1f(v.y);
    v.z = v.z > 0.f ? v.z : expm1f(v.z);
    v.w = v.w > 0.f ? v.w : expm1f(v.w);
  }
  out[i] = v;
  if (outbf) outbf[i] = make_ushort4(f2bf(v.x), f2bf(v.y), f2bf(v.z), f2bf(v.w));
}

// ---------- host orchestration ----------
extern "C" void kernel_launch(void* const* d_in, const int* in_sizes, int n_in,
                              void* d_out, int out_size, void* d_ws, size_t ws_size,
                              hipStream_t stream) {
  (void)in_sizes; (void)n_in; (void)out_size; (void)ws_size;
  const float*     x0  = (const float*)d_in[0];
  const float*     x1  = (const float*)d_in[1];
  const long long* src = (const long long*)d_in[2];
  const long long* dst = (const long long*)d_in[3];
  const float* fc0_W = (const float*)d_in[4];  const float* fc0_b = (const float*)d_in[5];
  const float* fc1_W = (const float*)d_in[6];  const float* fc1_b = (const float*)d_in[7];
  const float* W0  = (const float*)d_in[8];  const float* al0 = (const float*)d_in[9];
  const float* ar0 = (const float*)d_in[10]; const float* b0  = (const float*)d_in[11];
  const float* W1  = (const float*)d_in[12]; const float* al1 = (const float*)d_in[13];
  const float* ar1 = (const float*)d_in[14]; const float* b1  = (const float*)d_in[15];
  const float* W2  = (const float*)d_in[16]; const float* al2 = (const float*)d_in[17];
  const float* ar2 = (const float*)d_in[18]; const float* b2  = (const float*)d_in[19];
  const float* resW2 = (const float*)d_in[20];

  char* ws = (char*)d_ws;
  size_t off = 0;
  auto take = [&](size_t bytes) -> char* {
    char* p = ws + off; off = (off + bytes + 255) & ~(size_t)255; return p;
  };
  float*    ft   = (float*)take(50000ull * 512 * 4);
  float*    agg  = (float*)take(50000ull * 512 * 4);
  float*    hres = (float*)take(50000ull * 512 * 4);   // h1 f32 residual; fc f32 dump
  uint16_t* hbf  = (uint16_t*)take(50000ull * 512 * 2);
  uint16_t* x0bf = (uint16_t*)take(25008ull * 256 * 2);
  uint16_t* x1bf = (uint16_t*)take(25008ull * 256 * 2);
  float*    el   = (float*)take(50000ull * 8 * 4);
  float*    er   = (float*)take(50000ull * 8 * 4);
  unsigned* mbuf = (unsigned*)take(50000ull * 8 * 4);
  float*    sbuf = (float*)take(50000ull * 8 * 4);
  float*    ebuf = (float*)take(400000ull * 8 * 4);
  float*    res2 = (float*)take(50000ull * 16 * 4);
  uint16_t* fc0t = (uint16_t*)take(64ull * 256 * 2);
  uint16_t* fc1t = (uint16_t*)take(64ull * 256 * 2);
  uint16_t* W0t  = (uint16_t*)take(512ull * 64 * 2);
  uint16_t* W1t  = (uint16_t*)take(512ull * 512 * 2);
  uint16_t* W2t  = (uint16_t*)take(16ull * 512 * 2);
  uint16_t* rW2t = (uint16_t*)take(16ull * 512 * 2);

  // input + weight conversion (bf16; weights transposed to [N,K])
  long long nx4 = 25000ll * 256 / 4;
  k_cvt_bf16_4<<<(int)((nx4 + 255) / 256), 256, 0, stream>>>((const float4*)x0, (ushort4*)x0bf, nx4);
  k_cvt_bf16_4<<<(int)((nx4 + 255) / 256), 256, 0, stream>>>((const float4*)x1, (ushort4*)x1bf, nx4);
  k_transpose_bf16<<<(256 * 64 + 255) / 256, 256, 0, stream>>>(fc0_W, fc0t, 256, 64);
  k_transpose_bf16<<<(256 * 64 + 255) / 256, 256, 0, stream>>>(fc1_W, fc1t, 256, 64);
  k_transpose_bf16<<<(64 * 512 + 255) / 256, 256, 0, stream>>>(W0, W0t, 64, 512);
  k_transpose_bf16<<<(512 * 512 + 255) / 256, 256, 0, stream>>>(W1, W1t, 512, 512);
  k_transpose_bf16<<<(512 * 16 + 255) / 256, 256, 0, stream>>>(W2, W2t, 512, 16);
  k_transpose_bf16<<<(512 * 16 + 255) / 256, 256, 0, stream>>>(resW2, rW2t, 512, 16);

  auto gemm = [&](const uint16_t* A, const uint16_t* Bt, float* C, uint16_t* Cbf,
                  const float* bias, int M, int N, int K) {
    int tilesM = (M + 15) / 16;
    dim3 g((tilesM + 7) / 8, N / 64);
    k_gemm_lds<<<g, 256, 0, stream>>>(A, Bt, C, Cbf, bias, M, N, K, tilesM);
  };

  // fc projections -> h0 bf16 in hbf (f32 side dead, dumped into hres scratch)
  gemm(x0bf, fc0t, hres, hbf, fc0_b, 25000, 64, 256);
  gemm(x1bf, fc1t, hres + 25000ll * 64, hbf + 25000ll * 64, fc1_b, 25000, 64, 256);

  auto run_attn = [&](const float* ftp, const float* al, const float* ar, int H, int D) {
    int NH = NN * H;
    k_scores<<<(NH + 255) / 256, 256, 0, stream>>>(ftp, al, ar, el, er, NN, H, D);
    k_init_attn<<<(NH + 255) / 256, 256, 0, stream>>>(mbuf, sbuf, NH);
    long long nagg4 = (long long)NN * H * D / 4;
    k_zero_f32_4<<<(int)((nagg4 + 255) / 256), 256, 0, stream>>>((float4*)agg, nagg4);
    k_edge_max<<<(NE + 255) / 256, 256, 0, stream>>>(src, dst, el, er, ebuf, mbuf, NE, H);
    k_edge_exp<<<(NE + 255) / 256, 256, 0, stream>>>(dst, ebuf, mbuf, sbuf, NE, H);
    long long tot4 = (long long)NE * H * D / 4;
    k_edge_aggr<<<(int)((tot4 + 255) / 256), 256, 0, stream>>>(src, dst, ftp, ebuf, sbuf, agg,
                                                               tot4, H, D);
  };

  // ---- layer 0: 64 -> 8x64, ELU ----
  gemm(hbf, W0t, ft, nullptr, nullptr, NN, 512, 64);
  run_attn(ft, al0, ar0, 8, 64);
  k_finalize4<<<(int)((50000ll * 128 + 255) / 256), 256, 0, stream>>>(
      (const float4*)agg, nullptr, (const float4*)b0, (float4*)hres, (ushort4*)hbf,
      50000ll * 128, 128, 1);

  // ---- layer 1: 512 -> 8x64, identity residual, ELU; output = encoded ----
  gemm(hbf, W1t, ft, nullptr, nullptr, NN, 512, 512);
  run_attn(ft, al1, ar1, 8, 64);
  float* encoded = (float*)d_out + 50000ll * 16;
  k_finalize4<<<(int)((50000ll * 128 + 255) / 256), 256, 0, stream>>>(
      (const float4*)agg, (const float4*)hres, (const float4*)b1, (float4*)encoded,
      (ushort4*)hbf, 50000ll * 128, 128, 1);

  // ---- layer 2: 512 -> 1x16, linear residual, no act ----
  {
    int tilesM = (NN + 15) / 16;
    k_gemm_n16<<<(tilesM + 7) / 8, 256, 0, stream>>>(hbf, W2t, ft, NN, 512, tilesM);
    k_gemm_n16<<<(tilesM + 7) / 8, 256, 0, stream>>>(hbf, rW2t, res2, NN, 512, tilesM);
  }
  run_attn(ft, al2, ar2, 1, 16);
  k_finalize4<<<(int)((50000ll * 4 + 255) / 256), 256, 0, stream>>>(
      (const float4*)agg, (const float4*)res2, (const float4*)b2, (float4*)d_out,
      nullptr, 50000ll * 4, 4, 0);
}
